// GABTemplateMLP_66975720014055
// MI455X (gfx1250) — matrix operand order repaired and checked by reading, not compiled, not run on hardware
//
#include <hip/hip_runtime.h>
#include <math.h>

typedef __attribute__((ext_vector_type(16))) _Float16 v16h;
typedef __attribute__((ext_vector_type(8)))  float    v8f;

#define NPOS   32
#define SEQ    1024           // 32*32
#define NUMF   16
#define FIN    128            // 8*NUMF
#define HIDDEN 256
#define TOUT   32
#define NDIAG  63             // 2*NPOS-1
#define NTAB   (NDIAG*NDIAG)  // 3969 unique (dr,dc) rows
#define MPAD   4096           // padded row count (multiple of 16)
#define WAVES_PER_BLOCK 4

// A-matrix 16-bit 16x32 per-lane K index for vector element e (0..15):
// lanes 0-15 hold K in {0..15}, lanes 16-31 hold K in {16..31}
__device__ __forceinline__ int kmap(int e, int laneHi) {
    return e + laneHi * 16;
}

// ---------------------------------------------------------------------------
// Kernel 1: build the 4096x32 output table with WMMA.
// One wave per 16-row tile. GEMM1: feats(16x128) x W1^T(128x256) -> SiLU
// GEMM2: H(16x256) x W2^T(256x32), +b2, * 1/sqrt(32)
// ---------------------------------------------------------------------------
__global__ __launch_bounds__(128) void mlp_table_kernel(
    const float* __restrict__ freqs,
    const float* __restrict__ W1, const float* __restrict__ b1,
    const float* __restrict__ W2, const float* __restrict__ b2,
    float* __restrict__ table)
{
    __shared__ _Float16 Hs[WAVES_PER_BLOCK][16][HIDDEN];  // 32 KB
    __shared__ float sFreq[NUMF];

    const int tid    = threadIdx.x;
    const int lane   = tid & 31;
    const int wave   = tid >> 5;
    const int laneHi = lane >> 4;   // 0 or 1
    const int col16  = lane & 15;

    if (tid < NUMF) sFreq[tid] = freqs[tid];
    __syncthreads();

    const int tile = blockIdx.x * WAVES_PER_BLOCK + wave;   // 0..255

    // This lane supplies A-matrix row M = col16 of the tile.
    int row  = tile * 16 + col16;
    int rowc = (row < NTAB) ? row : 0;                      // pad rows duplicate row 0
    const float fdr = (float)(rowc / NDIAG - (NPOS - 1));   // dr in [-31,31]
    const float fdc = (float)(rowc % NDIAG - (NPOS - 1));   // dc in [-31,31]
    const float base[4] = { fdr, fdc, fdr + fdc, fdr - fdc };

    // --- Build feature A-fragments: 4 chunks of 16x32 f16 ---
    v16h afr[4];
    #pragma unroll
    for (int kc = 0; kc < 4; ++kc) {
        #pragma unroll
        for (int e = 0; e < 16; ++e) {
            int f = kc * 32 + kmap(e, laneHi);  // feature index 0..127
            int g = f >> 4;                     // group: sin/cos of dr,dc,dr+dc,dr-dc
            float arg = base[g >> 1] * sFreq[f & 15];
            float v = (g & 1) ? cosf(arg) : sinf(arg);
            afr[kc][e] = (_Float16)v;
        }
    }

    // --- GEMM1: 16 N-tiles over HIDDEN, K-loop of 4 WMMAs each ---
    #pragma unroll 1
    for (int n = 0; n < HIDDEN / 16; ++n) {
        v8f acc = {};
        #pragma unroll
        for (int kc = 0; kc < 4; ++kc) {
            // B[k][ncol] = W1[(n*16+ncol)*128 + k]; lane holds ncol=col16,
            // K = kc*32 + laneHi*16 + e  -> 16 contiguous floats
            const float4* bp4 = (const float4*)(W1 + (n * 16 + col16) * FIN
                                                + kc * 32 + laneHi * 16);
            v16h bfr;
            #pragma unroll
            for (int e4 = 0; e4 < 4; ++e4) {
                float4 w = bp4[e4];
                bfr[e4 * 4 + 0] = (_Float16)w.x;
                bfr[e4 * 4 + 1] = (_Float16)w.y;
                bfr[e4 * 4 + 2] = (_Float16)w.z;
                bfr[e4 * 4 + 3] = (_Float16)w.w;
            }
            acc = __builtin_amdgcn_wmma_f32_16x16x32_f16(
                false, afr[kc], false, bfr, (short)0, acc, false, false);
        }
        // C/D layout: lane col = col16, row = r + 8*laneHi
        float bb = b1[n * 16 + col16];
        #pragma unroll
        for (int r = 0; r < 8; ++r) {
            float x = acc[r] + bb;
            float h = x / (1.0f + expf(-x));    // SiLU
            Hs[wave][r + 8 * laneHi][n * 16 + col16] = (_Float16)h;
        }
    }
    __syncthreads();

    // --- Reload H in A-fragment layout: 8 chunks of 16x32 ---
    v16h hfr[8];
    #pragma unroll
    for (int kc = 0; kc < 8; ++kc) {
        #pragma unroll
        for (int e = 0; e < 16; ++e)
            hfr[kc][e] = Hs[wave][col16][kc * 32 + kmap(e, laneHi)];
    }

    // --- GEMM2: 2 N-tiles over TOUT=32, K-loop of 8 WMMAs ---
    const float inv = 0.17677669529663689f;     // 1/sqrt(32)
    #pragma unroll
    for (int n2 = 0; n2 < 2; ++n2) {
        v8f acc = {};
        #pragma unroll
        for (int kc = 0; kc < 8; ++kc) {
            const float4* bp4 = (const float4*)(W2 + (n2 * 16 + col16) * HIDDEN
                                                + kc * 32 + laneHi * 16);
            v16h bfr;
            #pragma unroll
            for (int e4 = 0; e4 < 4; ++e4) {
                float4 w = bp4[e4];
                bfr[e4 * 4 + 0] = (_Float16)w.x;
                bfr[e4 * 4 + 1] = (_Float16)w.y;
                bfr[e4 * 4 + 2] = (_Float16)w.z;
                bfr[e4 * 4 + 3] = (_Float16)w.w;
            }
            acc = __builtin_amdgcn_wmma_f32_16x16x32_f16(
                false, hfr[kc], false, bfr, (short)0, acc, false, false);
        }
        float bb = b2[n2 * 16 + col16];
        #pragma unroll
        for (int r = 0; r < 8; ++r) {
            int orow = tile * 16 + r + 8 * laneHi;
            table[orow * TOUT + n2 * 16 + col16] = (acc[r] + bb) * inv;
        }
    }
}

// ---------------------------------------------------------------------------
// Kernel 2: broadcast table -> 128 MB output. 8 threads per (i,j) pair,
// one float4 each; fully coalesced stores (the HBM-bound part: ~5.5us floor).
// ---------------------------------------------------------------------------
__global__ __launch_bounds__(256) void broadcast_kernel(
    const float* __restrict__ table, float* __restrict__ out)
{
    unsigned gid = blockIdx.x * 256u + threadIdx.x;   // 0 .. 8388607
    unsigned p = gid >> 3;                            // pair index i*1024+j
    unsigned q = gid & 7;                             // float4 slot within 32 outs
    int i = (int)(p >> 10), j = (int)(p & 1023);
    int d = ((i >> 5) - (j >> 5) + (NPOS - 1)) * NDIAG
          + ((i & 31) - (j & 31) + (NPOS - 1));
    float4 v = ((const float4*)table)[d * 8 + (int)q];
    ((float4*)out)[p * 8 + q] = v;
}

extern "C" void kernel_launch(void* const* d_in, const int* in_sizes, int n_in,
                              void* d_out, int out_size, void* d_ws, size_t ws_size,
                              hipStream_t stream) {
    // inputs: 0:seq_len 1:freqs 2:W1 3:b1 4:W2 5:b2 6:dr 7:dc
    const float* freqs = (const float*)d_in[1];
    const float* W1    = (const float*)d_in[2];
    const float* b1    = (const float*)d_in[3];
    const float* W2    = (const float*)d_in[4];
    const float* b2    = (const float*)d_in[5];
    float* table = (float*)d_ws;            // needs MPAD*TOUT*4 = 512 KB scratch
    float* out   = (float*)d_out;

    mlp_table_kernel<<<MPAD / 16 / WAVES_PER_BLOCK, 128, 0, stream>>>(
        freqs, W1, b1, W2, b2, table);
    broadcast_kernel<<<(SEQ * SEQ * 8) / 256, 256, 0, stream>>>(table, out);
}
